// NGCNLin_6098853560423
// MI455X (gfx1250) — matrix operand order repaired and checked by reading, not compiled, not run on hardware
//
#include <hip/hip_runtime.h>
#include <math.h>

#define N_NODES   8192
#define N_EDGES   65536
#define IN_DIM    768
#define HIDDEN    4096
#define N_CLASSES 16

typedef __attribute__((ext_vector_type(8)))  float          v8f;
typedef __attribute__((ext_vector_type(16))) __bf16         v16bf;
typedef __attribute__((ext_vector_type(8)))  unsigned short v8us;
typedef __attribute__((ext_vector_type(4)))  unsigned short v4us;
typedef __attribute__((ext_vector_type(4)))  int            v4i;

union FragBF {
    struct { v8us lo, hi; } u;
    v16bf v;
};

// Async global->LDS staging (CDNA5): guarded so the file compiles on either
// toolchain; if absent we fall back to global_load_b128 + ds_store_b128.
#if defined(__gfx1250__) && \
    __has_builtin(__builtin_amdgcn_global_load_async_to_lds_b128) && \
    __has_builtin(__builtin_amdgcn_s_wait_asynccnt)
#define HAVE_ASYNC_LDS 1
#else
#define HAVE_ASYNC_LDS 0
#endif

#if HAVE_ASYNC_LDS
// builtin signature: (int4 AS1*, int4 AS3*, imm offset, imm cpol)
typedef __attribute__((address_space(1))) v4i* gv4i_p;
typedef __attribute__((address_space(3))) v4i* lv4i_p;
// per-lane: copy 16B from global to LDS, tracked by ASYNCcnt
#define ASYNC_CP16(gp, lp)                                                  \
    __builtin_amdgcn_global_load_async_to_lds_b128(                         \
        (gv4i_p)(unsigned short*)(gp), (lv4i_p)(lp), 0, 0)
#endif

__device__ __forceinline__ unsigned short f2bf(float f) {
    unsigned int u = __float_as_uint(f);
    u += 0x7fffu + ((u >> 16) & 1u);   // round-to-nearest-even
    return (unsigned short)(u >> 16);
}
__device__ __forceinline__ float bf2f(unsigned short h) {
    return __uint_as_float(((unsigned int)h) << 16);
}

// ---------------------------------------------------------------- converts
__global__ void conv_bf16_kernel(const float* __restrict__ in,
                                 unsigned short* __restrict__ out, int n) {
    int i = blockIdx.x * blockDim.x + threadIdx.x;
    if (i < n) out[i] = f2bf(in[i]);
}

// in: [K,N] row-major f32 -> out: [N,K] row-major bf16 (i.e., column-major)
__global__ void conv_bf16_transpose_kernel(const float* __restrict__ in,
                                           unsigned short* __restrict__ out,
                                           int K, int N) {
    int i = blockIdx.x * blockDim.x + threadIdx.x;
    if (i >= K * N) return;
    int k = i / N, n = i % N;
    out[(size_t)n * K + k] = f2bf(in[i]);
}

// ---------------------------------------------------------------- degrees
__global__ void deg_kernel(const long long* __restrict__ dst,
                           float* __restrict__ deg, int E) {
    int e = blockIdx.x * blockDim.x + threadIdx.x;
    if (e < E) atomicAdd(&deg[(int)dst[e]], 1.0f);
}

__global__ void dinv_kernel(float* __restrict__ deg_dinv, int n) {
    int i = blockIdx.x * blockDim.x + threadIdx.x;
    if (i < n) deg_dinv[i] = rsqrtf(deg_dinv[i] + 1.0f);  // +1 = self loop
}

// ---------------------------------------------------------------- GEMM1
// Hbf[8192,4096](bf16) = Xbf[8192,768](bf16) @ W1t^T  (W1t is [4096,768] col-major copy)
// Block: 128 threads (4 waves), 64x128 output tile, K-step 32.
// Each wave: 16 rows x 128 cols = 8 WMMA accumulators per A-fragment.
#define BM 64
#define BN 128
#define BK 32
#define NJ (BN / 16)   // 8 n-tiles per wave
#define LDT 40         // padded LDS stride (elements) -> 80B rows, 16B aligned

__global__ __launch_bounds__(128) void gemm1_wmma_kernel(
        const unsigned short* __restrict__ Xbf,
        const unsigned short* __restrict__ W1t,
        unsigned short* __restrict__ Hbf) {
    __shared__ unsigned short As[BM * LDT];  // [row][k]
    __shared__ unsigned short Bs[BN * LDT];  // [col][k]

    const int n0   = blockIdx.x * BN;
    const int m0   = blockIdx.y * BM;
    const int tid  = threadIdx.x;
    const int lane = tid & 31;
    const int wid  = tid >> 5;      // 0..3 -> rows wid*16..wid*16+15
    const int l16  = lane & 15;
    const int lh   = lane >> 4;     // half-wave select

    v8f acc[NJ] = {};

    for (int k0 = 0; k0 < IN_DIM; k0 += BK) {
        // ---- stage A tile: 64 rows x 32 k (256 x 16B chunks, 2 per thread)
        #pragma unroll
        for (int ci = 0; ci < 2; ++ci) {
            int id  = tid * 2 + ci;         // [0,256)
            int row = id >> 2, q = id & 3;  // q: 8-element chunk in k
            const unsigned short* g = Xbf + (size_t)(m0 + row) * IN_DIM + k0 + q * 8;
            unsigned short* l = As + row * LDT + q * 8;
#if HAVE_ASYNC_LDS
            ASYNC_CP16(g, l);
#else
            *(v8us*)l = *(const v8us*)g;
#endif
        }
        // ---- stage B tile: 128 cols x 32 k (512 x 16B chunks, 4 per thread)
        #pragma unroll
        for (int q = 0; q < 4; ++q) {
            const unsigned short* g = W1t + (size_t)(n0 + tid) * IN_DIM + k0 + q * 8;
            unsigned short* l = Bs + tid * LDT + q * 8;
#if HAVE_ASYNC_LDS
            ASYNC_CP16(g, l);
#else
            *(v8us*)l = *(const v8us*)g;
#endif
        }
#if HAVE_ASYNC_LDS
        __builtin_amdgcn_s_wait_asynccnt(0);
#endif
        __syncthreads();

        // A fragment: lane holds row (wid*16+l16), K chunks [lh*8,+8) and [16+lh*8,+8)
        FragBF af;
        {
            const unsigned short* p = As + (wid * 16 + l16) * LDT + lh * 8;
            af.u.lo = *(const v8us*)(p);
            af.u.hi = *(const v8us*)(p + 16);
        }
        #pragma unroll
        for (int j = 0; j < NJ; ++j) {
            // B fragment: lane holds col (j*16+l16), K chunks [lh*8,+8) and [16+lh*8,+8)
            FragBF bf;
            const unsigned short* p = Bs + (j * 16 + l16) * LDT + lh * 8;
            bf.u.lo = *(const v8us*)(p);
            bf.u.hi = *(const v8us*)(p + 16);
            acc[j] = __builtin_amdgcn_wmma_f32_16x16x32_bf16(
                false, af.v, false, bf.v, (short)0, acc[j], false, false);
        }
        __syncthreads();
    }

    // C/D layout: VGPR v, lanes<16 -> M=v, lanes>=16 -> M=v+8; N=l16
    #pragma unroll
    for (int j = 0; j < NJ; ++j) {
        #pragma unroll
        for (int v = 0; v < 8; ++v) {
            int row = m0 + wid * 16 + v + lh * 8;
            int col = n0 + j * 16 + l16;
            Hbf[(size_t)row * HIDDEN + col] = f2bf(acc[j][v]);
        }
    }
}

// ---------------------------------------------------------------- aggregation
// agg[dst] += norm * h[src]   (edges + self loops), f32 atomics, bf16 gather
__global__ __launch_bounds__(256) void aggregate_kernel(
        const unsigned short* __restrict__ Hbf,
        const long long* __restrict__ src,
        const long long* __restrict__ dst,
        const float* __restrict__ dinv,
        float* __restrict__ agg, int E) {
    int e = blockIdx.x;
    int s, d;
    if (e < E) { s = (int)src[e]; d = (int)dst[e]; }
    else       { s = d = e - E; }                    // self loop
    float nrm = dinv[s] * dinv[d];
    int f0 = blockIdx.y * 1024 + threadIdx.x * 4;
    v4us h = *(const v4us*)(Hbf + (size_t)s * HIDDEN + f0);
    float* ap = agg + (size_t)d * HIDDEN + f0;
    atomicAdd(ap + 0, nrm * bf2f(h[0]));
    atomicAdd(ap + 1, nrm * bf2f(h[1]));
    atomicAdd(ap + 2, nrm * bf2f(h[2]));
    atomicAdd(ap + 3, nrm * bf2f(h[3]));
}

// ---------------------------------------------------------------- relu+bias -> bf16
__global__ void relu_bias_kernel(const float* __restrict__ agg,
                                 const float* __restrict__ b1,
                                 unsigned short* __restrict__ out) {
    size_t idx = (size_t)blockIdx.x * blockDim.x + threadIdx.x;
    int f = (int)(idx & (HIDDEN - 1));
    float v = agg[idx] + b1[f];
    out[idx] = f2bf(fmaxf(v, 0.0f));
}

// ---------------------------------------------------------------- GEMM2
// logits[8192,16] = Hrelu(bf16) @ W2 + b2 ; W2t is [16,4096] col-major bf16 copy.
// One wave computes a full 16x16 tile; K loop = 128 WMMAs.
__global__ __launch_bounds__(128) void gemm2_wmma_kernel(
        const unsigned short* __restrict__ Hrelu,
        const unsigned short* __restrict__ W2t,
        const float* __restrict__ b2,
        float* __restrict__ logits) {
    const int tid  = threadIdx.x;
    const int lane = tid & 31;
    const int wid  = tid >> 5;
    const int l16  = lane & 15;
    const int lh   = lane >> 4;
    const int m0   = blockIdx.x * 64 + wid * 16;

    v8f acc = {};
    for (int k0 = 0; k0 < HIDDEN; k0 += 32) {
        FragBF a, b;
        const unsigned short* pa = Hrelu + (size_t)(m0 + l16) * HIDDEN + k0 + lh * 8;
        a.u.lo = *(const v8us*)(pa);
        a.u.hi = *(const v8us*)(pa + 16);
        const unsigned short* pb = W2t + (size_t)l16 * HIDDEN + k0 + lh * 8;
        b.u.lo = *(const v8us*)(pb);
        b.u.hi = *(const v8us*)(pb + 16);
        acc = __builtin_amdgcn_wmma_f32_16x16x32_bf16(
            false, a.v, false, b.v, (short)0, acc, false, false);
    }
    #pragma unroll
    for (int v = 0; v < 8; ++v) {
        int row = m0 + v + lh * 8;
        int col = l16;
        logits[row * N_CLASSES + col] = acc[v] + b2[col];
    }
}

// ---------------------------------------------------------------- log_softmax
__global__ void logsoftmax_kernel(const float* __restrict__ logits,
                                  float* __restrict__ out, int M) {
    int r = blockIdx.x * blockDim.x + threadIdx.x;
    if (r >= M) return;
    float v[N_CLASSES];
    float m = -1e30f;
    #pragma unroll
    for (int c = 0; c < N_CLASSES; ++c) {
        v[c] = logits[r * N_CLASSES + c];
        m = fmaxf(m, v[c]);
    }
    float s = 0.0f;
    #pragma unroll
    for (int c = 0; c < N_CLASSES; ++c) s += __expf(v[c] - m);
    float ls = __logf(s);
    #pragma unroll
    for (int c = 0; c < N_CLASSES; ++c) out[r * N_CLASSES + c] = v[c] - m - ls;
}

// ================================================================ launch
extern "C" void kernel_launch(void* const* d_in, const int* in_sizes, int n_in,
                              void* d_out, int out_size, void* d_ws, size_t ws_size,
                              hipStream_t stream) {
    const float*     x   = (const float*)d_in[0];
    const long long* ei  = (const long long*)d_in[1];   // [2, E] int64
    const float*     W1  = (const float*)d_in[2];
    const float*     b1  = (const float*)d_in[3];
    const float*     W2  = (const float*)d_in[4];
    const float*     b2  = (const float*)d_in[5];
    float*           out = (float*)d_out;

    const long long* src = ei;
    const long long* dst = ei + N_EDGES;

    // workspace carve-up (256B aligned)
    char* base = (char*)d_ws;
    size_t off = 0;
    auto carve = [&](size_t bytes) {
        size_t o = off;
        off += (bytes + 255) & ~(size_t)255;
        return o;
    };
    size_t o_hbf    = carve((size_t)N_NODES * HIDDEN * 2);      // h bf16 (reused for relu out)
    size_t o_agg    = carve((size_t)N_NODES * HIDDEN * 4);      // agg f32 (atomic target)
    size_t o_xbf    = carve((size_t)N_NODES * IN_DIM * 2);      // x bf16
    size_t o_w1t    = carve((size_t)HIDDEN * IN_DIM * 2);       // W1^T bf16
    size_t o_w2t    = carve((size_t)N_CLASSES * HIDDEN * 2);    // W2^T bf16
    size_t o_deg    = carve((size_t)N_NODES * 4);               // deg->dinv f32
    size_t o_logits = carve((size_t)N_NODES * N_CLASSES * 4);   // logits f32
    (void)ws_size;

    unsigned short* hbf    = (unsigned short*)(base + o_hbf);
    float*          agg    = (float*)(base + o_agg);
    unsigned short* xbf    = (unsigned short*)(base + o_xbf);
    unsigned short* w1t    = (unsigned short*)(base + o_w1t);
    unsigned short* w2t    = (unsigned short*)(base + o_w2t);
    float*          dinv   = (float*)(base + o_deg);
    float*          logits = (float*)(base + o_logits);

    // zero atomic targets (graph-capturable)
    (void)hipMemsetAsync(agg,  0, (size_t)N_NODES * HIDDEN * 4, stream);
    (void)hipMemsetAsync(dinv, 0, (size_t)N_NODES * 4, stream);

    // precision conversion / transposition
    {
        int n = N_NODES * IN_DIM;
        conv_bf16_kernel<<<(n + 255) / 256, 256, 0, stream>>>(x, xbf, n);
    }
    {
        int n = IN_DIM * HIDDEN;
        conv_bf16_transpose_kernel<<<(n + 255) / 256, 256, 0, stream>>>(W1, w1t, IN_DIM, HIDDEN);
    }
    {
        int n = HIDDEN * N_CLASSES;
        conv_bf16_transpose_kernel<<<(n + 255) / 256, 256, 0, stream>>>(W2, w2t, HIDDEN, N_CLASSES);
    }

    // degree + dinv
    deg_kernel<<<(N_EDGES + 255) / 256, 256, 0, stream>>>(dst, dinv, N_EDGES);
    dinv_kernel<<<(N_NODES + 255) / 256, 256, 0, stream>>>(dinv, N_NODES);

    // GEMM1: h = x @ W1  (bf16 WMMA)
    {
        dim3 grid(HIDDEN / BN, N_NODES / BM);   // 32 x 128
        gemm1_wmma_kernel<<<grid, 128, 0, stream>>>(xbf, w1t, hbf);
    }

    // aggregation (edges + self loops)
    {
        dim3 grid(N_EDGES + N_NODES, HIDDEN / 1024);  // 73728 x 4
        aggregate_kernel<<<grid, 256, 0, stream>>>(hbf, src, dst, dinv, agg, N_EDGES);
    }

    // relu(agg + b1) -> bf16, overwriting h buffer
    {
        size_t n = (size_t)N_NODES * HIDDEN;
        relu_bias_kernel<<<(unsigned)(n / 256), 256, 0, stream>>>(agg, b1, hbf);
    }

    // GEMM2: logits = relu_h @ W2 + b2  (bf16 WMMA, one wave per 16x16 tile)
    gemm2_wmma_kernel<<<N_NODES / 64, 128, 0, stream>>>(hbf, w2t, b2, logits);

    // log_softmax -> output
    logsoftmax_kernel<<<(N_NODES + 255) / 256, 256, 0, stream>>>(logits, out, N_NODES);

    (void)in_sizes; (void)n_in; (void)out_size;
}
